// ConstraintBeamSearch_28509992911464
// MI455X (gfx1250) — matrix operand order repaired and checked by reading, not compiled, not run on hardware
//
#include <hip/hip_runtime.h>
#include <math.h>

// ---------------- problem constants (from reference) ----------------
#define END_INDEX 0
constexpr int Tn   = 20;      // MAX_STEPS
constexpr int BEAMW = 5;      // BEAM
constexpr int KSEL = 5;       // per_node_beam_size
constexpr int Bn = 32, Sn = 4, Vn = 20000, Hn = 512, Dn = 2048;
constexpr int SB   = Sn * BEAMW;   // 20
constexpr int NSEQ = Bn * SB;      // 640
#define VNF   (-1e20f)
#define NEGINF (-3.0e38f)

typedef __bf16 bf16;
typedef __attribute__((ext_vector_type(16))) __bf16 v16bf;
typedef __attribute__((ext_vector_type(8)))  float  v8f;
typedef __attribute__((ext_vector_type(4)))  unsigned u32x4;

// float -> bf16 round-to-nearest-even
__device__ __forceinline__ bf16 f2bf(float f) {
    unsigned u = __builtin_bit_cast(unsigned, f);
    u += 0x7FFFu + ((u >> 16) & 1u);
    unsigned short s = (unsigned short)(u >> 16);
    return __builtin_bit_cast(bf16, s);
}

// ---------------- fp32 -> bf16 conversion (row-major copy) ----------------
__global__ void cvt_bf16_kernel(const float* __restrict__ src, bf16* __restrict__ dst, long n) {
    long i = (long)blockIdx.x * blockDim.x + threadIdx.x;
    if (i < n) dst[i] = f2bf(src[i]);
}

// ---------------- fp32 -> bf16 transposing conversion ----------------
// src[R][C] row-major  ->  dst[C][R] row-major (i.e. dst = src^T), via LDS 32x32 tile
__global__ void __launch_bounds__(256) cvt_bf16_transpose_kernel(
        const float* __restrict__ src, bf16* __restrict__ dst, int R, int C) {
    __shared__ float tile[32][33];
    const int c0 = blockIdx.x * 32;
    const int r0 = blockIdx.y * 32;
    const int tx = threadIdx.x & 31;
    const int ty = threadIdx.x >> 5;   // 0..7
    for (int i = ty; i < 32; i += 8) {
        int r = r0 + i, c = c0 + tx;
        tile[i][tx] = (r < R && c < C) ? src[(long)r * C + c] : 0.0f;
    }
    __syncthreads();
    for (int i = ty; i < 32; i += 8) {
        int c = c0 + i, r = r0 + tx;
        if (c < C && r < R) dst[(long)c * R + r] = f2bf(tile[tx][i]);
    }
}

// ---------------- WMMA bf16 GEMM: C[M,N] = A[M,K] * B[K,N], B given TRANSPOSED ----------------
// Bt is [N][K] row-major. One wave32 computes a 16x32 strip (2 accumulators) so the
// A fragment is reused and both fragments load as contiguous global_load_b128 pairs.
__global__ void __launch_bounds__(256) gemm_bf16_wmma_kernel(
        const bf16* __restrict__ A, const bf16* __restrict__ Bt, float* __restrict__ C,
        int M, int K, int N) {
    const int lane   = threadIdx.x & 31;
    const int wave   = blockIdx.x * (blockDim.x >> 5) + (threadIdx.x >> 5);
    const int strips = N >> 5;                 // 32-column strips
    const int tileM  = wave / strips;
    const int strip  = wave % strips;
    if (tileM >= (M >> 4)) return;             // wave-uniform exit: EXEC all-ones for WMMA

    const int mrow = (tileM << 4) + (lane & 15);
    const int n0   = (strip << 5) + (lane & 15);
    const int n1   = n0 + 16;
    const int aKb  = (lane < 16) ? 0 : 16;     // A 16x32 fragment K-base per half-wave
    const int bKb  = (lane < 16) ? 0 : 16;     // B 32x16 fragment K-base per half-wave
    const bf16* Ap  = A  + (long)mrow * K;
    const bf16* Bp0 = Bt + (long)n0 * K;
    const bf16* Bp1 = Bt + (long)n1 * K;

    v8f acc0 = {}, acc1 = {};
#pragma unroll 4
    for (int k0 = 0; k0 < K; k0 += 32) {
        union { v16bf v; u32x4 q[2]; } a, b0, b1;
        a.q[0]  = *reinterpret_cast<const u32x4*>(Ap  + k0 + aKb);
        a.q[1]  = *reinterpret_cast<const u32x4*>(Ap  + k0 + aKb + 8);
        b0.q[0] = *reinterpret_cast<const u32x4*>(Bp0 + k0 + bKb);
        b0.q[1] = *reinterpret_cast<const u32x4*>(Bp0 + k0 + bKb + 8);
        b1.q[0] = *reinterpret_cast<const u32x4*>(Bp1 + k0 + bKb);
        b1.q[1] = *reinterpret_cast<const u32x4*>(Bp1 + k0 + bKb + 8);
        if (k0 + 32 < K) {
            __builtin_prefetch(Ap  + k0 + 32, 0, 0);   // global_prefetch_b8
            __builtin_prefetch(Bp0 + k0 + 32, 0, 0);
            __builtin_prefetch(Bp1 + k0 + 32, 0, 0);
        }
        acc0 = __builtin_amdgcn_wmma_f32_16x16x32_bf16(
                   false, a.v, false, b0.v, (short)0, acc0, false, false);
        acc1 = __builtin_amdgcn_wmma_f32_16x16x32_bf16(
                   false, a.v, false, b1.v, (short)0, acc1, false, false);
    }
    const int cmBase = (tileM << 4) + ((lane < 16) ? 0 : 8);
#pragma unroll
    for (int r = 0; r < 8; ++r) {
        C[(long)(cmBase + r) * N + n0] = acc0[r];
        C[(long)(cmBase + r) * N + n1] = acc1[r];
    }
}

// ---------------- step-0 hidden: h0 = tanh(emb[start] + img_proj) ----------------
__global__ void step0_hidden_kernel(const float* __restrict__ emb, const float* __restrict__ imgProj,
                                    const int* __restrict__ start, float* __restrict__ h0,
                                    bf16* __restrict__ h0B) {
    int i = blockIdx.x * blockDim.x + threadIdx.x;
    if (i >= Bn * Hn) return;
    int b = i / Hn, c = i % Hn;
    float v = tanhf(emb[(long)start[b] * Hn + c] + imgProj[i]);
    h0[i] = v;
    h0B[i] = f2bf(v);
}

// ---------------- broadcast h0 over (S,BEAM) into bf16 h ----------------
__global__ void broadcast_h_kernel(const float* __restrict__ h0, bf16* __restrict__ hB) {
    long i = (long)blockIdx.x * blockDim.x + threadIdx.x;
    if (i >= (long)NSEQ * Hn) return;
    int row = (int)(i / Hn), c = (int)(i % Hn);
    int b = row / SB;
    hB[i] = f2bf(h0[b * Hn + c]);
}

// ---------------- RNN epilogue: hNew = tanh(pre + emb[tok] + imgProj) ----------------
__global__ void rnn_epilogue_kernel(const float* __restrict__ pre, const float* __restrict__ emb,
                                    const float* __restrict__ imgProj, const int* __restrict__ tokens,
                                    float* __restrict__ hNew, bf16* __restrict__ hNewB) {
    long i = (long)blockIdx.x * blockDim.x + threadIdx.x;
    if (i >= (long)NSEQ * Hn) return;
    int row = (int)(i / Hn), c = (int)(i % Hn);
    int b = row / SB;
    int tok = tokens[row];
    float v = tanhf(pre[i] + emb[(long)tok * Hn + c] + imgProj[b * Hn + c]);
    hNew[i] = v;
    hNewB[i] = f2bf(v);
}

// ---------------- row-wise log-softmax in place ----------------
__global__ void __launch_bounds__(256) log_softmax_kernel(float* __restrict__ x, int cols) {
    __shared__ float red[256];
    const int tid = threadIdx.x;
    float* xr = x + (long)blockIdx.x * cols;
    float m = NEGINF;
    for (int c = tid; c < cols; c += 256) m = fmaxf(m, xr[c]);
    red[tid] = m; __syncthreads();
    for (int s = 128; s > 0; s >>= 1) {
        if (tid < s) red[tid] = fmaxf(red[tid], red[tid + s]);
        __syncthreads();
    }
    m = red[0]; __syncthreads();
    float sum = 0.f;
    for (int c = tid; c < cols; c += 256) sum += expf(xr[c] - m);
    red[tid] = sum; __syncthreads();
    for (int s = 128; s > 0; s >>= 1) {
        if (tid < s) red[tid] += red[tid + s];
        __syncthreads();
    }
    float lse = m + logf(red[0]);
    for (int c = tid; c < cols; c += 256) xr[c] -= lse;
}

// ---------------- merge two descending top-5 lists ----------------
__device__ __forceinline__ void merge_top5(float* av, int* ai, const float* bv, const int* bi) {
    float rv[KSEL]; int ri[KSEL];
    int i = 0, j = 0;
#pragma unroll
    for (int o = 0; o < KSEL; ++o) {
        if (av[i] >= bv[j]) { rv[o] = av[i]; ri[o] = ai[i]; ++i; }
        else                { rv[o] = bv[j]; ri[o] = bi[j]; ++j; }
    }
#pragma unroll
    for (int o = 0; o < KSEL; ++o) { av[o] = rv[o]; ai[o] = ri[o]; }
}

__device__ __forceinline__ void block_top5(float val, int idx, float* lv, int* li) {
    if (val > lv[KSEL - 1]) {
        int e = KSEL - 1;
        while (e > 0 && val > lv[e - 1]) { lv[e] = lv[e - 1]; li[e] = li[e - 1]; --e; }
        lv[e] = val; li[e] = idx;
    }
}

// ---------------- initial masked top-BEAM (state 0 -> s) ----------------
__global__ void __launch_bounds__(256) init_top5_kernel(
        const float* __restrict__ lp0, const unsigned char* __restrict__ stt,
        float* __restrict__ lastLp, int* __restrict__ preds0) {
    const int blk = blockIdx.x;
    const int b = blk / Sn, st_t = blk % Sn;
    const int tid = threadIdx.x;
    const float* row = lp0 + (long)b * Vn;
    const unsigned char* mask = stt + (((long)b * Sn + 0) * Sn + st_t) * (long)Vn;

    float lv[KSEL]; int li[KSEL];
    for (int e = 0; e < KSEL; ++e) { lv[e] = NEGINF; li[e] = 0; }
    for (int v = tid; v < Vn; v += 256)
        block_top5(mask[v] ? row[v] : VNF, v, lv, li);

    __shared__ float sv[256 * KSEL];
    __shared__ int   si[256 * KSEL];
    for (int e = 0; e < KSEL; ++e) { sv[tid * KSEL + e] = lv[e]; si[tid * KSEL + e] = li[e]; }
    __syncthreads();
    for (int stride = 128; stride > 0; stride >>= 1) {
        if (tid < stride) {
            float a1[KSEL], b1[KSEL]; int a2[KSEL], b2[KSEL];
            for (int e = 0; e < KSEL; ++e) {
                a1[e] = sv[tid * KSEL + e];            a2[e] = si[tid * KSEL + e];
                b1[e] = sv[(tid + stride) * KSEL + e]; b2[e] = si[(tid + stride) * KSEL + e];
            }
            merge_top5(a1, a2, b1, b2);
            for (int e = 0; e < KSEL; ++e) { sv[tid * KSEL + e] = a1[e]; si[tid * KSEL + e] = a2[e]; }
        }
        __syncthreads();
    }
    if (tid < BEAMW) {
        lastLp[(b * Sn + st_t) * BEAMW + tid] = sv[tid];
        preds0[b * SB + st_t * BEAMW + tid]   = si[tid];
    }
}

// ---------------- per-step masked top-K over V for each (b, st, ss, beam) ----------------
__global__ void __launch_bounds__(256) masked_top5_kernel(
        const float* __restrict__ lp, const unsigned char* __restrict__ stt,
        const int* __restrict__ lastPreds, float* __restrict__ topLp, int* __restrict__ topCls) {
    const int blk = blockIdx.x;
    int q = blk;
    const int beam = q % BEAMW; q /= BEAMW;
    const int ss   = q % Sn;    q /= Sn;
    const int st_t = q % Sn;
    const int b    = q / Sn;
    const int tid = threadIdx.x;

    const float* row = lp + ((long)(b * Sn + ss) * BEAMW + beam) * Vn;
    // mask_t[b, st, ss, v] = state_transform[b, ss, st, v]
    const unsigned char* mask = stt + (((long)b * Sn + ss) * Sn + st_t) * (long)Vn;
    const bool fin = (lastPreds[b * SB + ss * BEAMW + beam] == END_INDEX);

    float lv[KSEL]; int li[KSEL];
    for (int e = 0; e < KSEL; ++e) { lv[e] = NEGINF; li[e] = 0; }
    for (int v = tid; v < Vn; v += 256) {
        float val = mask[v] ? (fin ? (v == END_INDEX ? 0.0f : VNF) : row[v]) : VNF;
        block_top5(val, v, lv, li);
    }
    __shared__ float sv[256 * KSEL];
    __shared__ int   si[256 * KSEL];
    for (int e = 0; e < KSEL; ++e) { sv[tid * KSEL + e] = lv[e]; si[tid * KSEL + e] = li[e]; }
    __syncthreads();
    for (int stride = 128; stride > 0; stride >>= 1) {
        if (tid < stride) {
            float a1[KSEL], b1[KSEL]; int a2[KSEL], b2[KSEL];
            for (int e = 0; e < KSEL; ++e) {
                a1[e] = sv[tid * KSEL + e];            a2[e] = si[tid * KSEL + e];
                b1[e] = sv[(tid + stride) * KSEL + e]; b2[e] = si[(tid + stride) * KSEL + e];
            }
            merge_top5(a1, a2, b1, b2);
            for (int e = 0; e < KSEL; ++e) { sv[tid * KSEL + e] = a1[e]; si[tid * KSEL + e] = a2[e]; }
        }
        __syncthreads();
    }
    if (tid < KSEL) {
        topLp[(long)blk * KSEL + tid]  = sv[tid];
        topCls[(long)blk * KSEL + tid] = si[tid];
    }
}

// ---------------- beam combine: top-BEAM over (S_src * BEAM * K) candidates ----------------
__global__ void __launch_bounds__(128) combine_kernel(
        const float* __restrict__ topLp, const int* __restrict__ topCls,
        const float* __restrict__ lastLpPrev, float* __restrict__ lastLpNext,
        int* __restrict__ predsT, int* __restrict__ bpsT) {
    const int blk = blockIdx.x;
    const int b = blk / Sn, st_t = blk % Sn;
    const int NC = Sn * BEAMW * KSEL;  // 100
    __shared__ float cand[Sn * BEAMW * KSEL];
    __shared__ int   cls[Sn * BEAMW * KSEL];
    const int tid = threadIdx.x;
    for (int idx = tid; idx < NC; idx += blockDim.x) {
        int k = idx % KSEL;
        int sb = idx / KSEL;
        int beam = sb % BEAMW, ss = sb / BEAMW;
        long rrow = (((long)b * Sn + st_t) * Sn + ss) * BEAMW + beam;
        cand[idx] = topLp[rrow * KSEL + k] + lastLpPrev[(b * Sn + ss) * BEAMW + beam];
        cls[idx]  = topCls[rrow * KSEL + k];
    }
    __syncthreads();
    if (tid == 0) {
        bool used[Sn * BEAMW * KSEL];
        for (int i = 0; i < NC; ++i) used[i] = false;
        for (int p = 0; p < BEAMW; ++p) {
            float bv = NEGINF; int bi = 0;
            for (int i = 0; i < NC; ++i)
                if (!used[i] && cand[i] > bv) { bv = cand[i]; bi = i; }
            used[bi] = true;
            lastLpNext[(b * Sn + st_t) * BEAMW + p] = bv;
            predsT[b * SB + st_t * BEAMW + p] = cls[bi];
            bpsT[b * SB + st_t * BEAMW + p]   = bi / KSEL;  // index into S_src*BEAM
        }
    }
}

// ---------------- gather hidden state through backpointers (f32 -> bf16) ----------------
__global__ void gather_h_kernel(const float* __restrict__ hNew, const int* __restrict__ bpsT,
                                bf16* __restrict__ hB) {
    const int r = blockIdx.x;
    const int b = r / SB, j = r % SB;
    const int src = b * SB + bpsT[b * SB + j];
    const float* sp = hNew + (long)src * Hn;
    bf16* db = hB + (long)r * Hn;
    for (int c = threadIdx.x; c < Hn; c += blockDim.x) db[c] = f2bf(sp[c]);
}

// ---------------- backtrack + emit outputs ----------------
// d_out layout: all_predictions [B, S, BEAM, T] as float, then last_lp [B, S, BEAM]
__global__ void backtrack_kernel(const int* __restrict__ preds, const int* __restrict__ bps,
                                 const float* __restrict__ lastLp, float* __restrict__ out) {
    int idx = blockIdx.x * blockDim.x + threadIdx.x;
    if (idx >= NSEQ) return;
    const int b = idx / SB, j = idx % SB;
    float* o = out + (long)idx * Tn;
    o[Tn - 1] = (float)preds[(Tn - 1) * NSEQ + b * SB + j];
    int cur = bps[(Tn - 1) * NSEQ + b * SB + j];
    for (int t = Tn - 2; t >= 1; --t) {
        o[t] = (float)preds[t * NSEQ + b * SB + cur];
        cur  = bps[t * NSEQ + b * SB + cur];
    }
    o[0] = (float)preds[0 * NSEQ + b * SB + cur];
    out[(long)NSEQ * Tn + idx] = lastLp[idx];
}

// ======================= host orchestration =======================
extern "C" void kernel_launch(void* const* d_in, const int* in_sizes, int n_in,
                              void* d_out, int out_size, void* d_ws, size_t ws_size,
                              hipStream_t stream) {
    (void)in_sizes; (void)n_in; (void)out_size; (void)ws_size;
    const float* image_feature      = (const float*)d_in[0];
    const int*   start_predictions  = (const int*)d_in[1];
    const unsigned char* state_tr   = (const unsigned char*)d_in[2];  // bool mask [B,S,S,V]
    const float* emb   = (const float*)d_in[3];
    const float* W_h   = (const float*)d_in[4];
    const float* W_im  = (const float*)d_in[5];
    const float* W_out = (const float*)d_in[6];
    float* out = (float*)d_out;

    // bump allocator over d_ws (256B aligned)
    char* p = (char*)d_ws;
    auto alloc = [&](size_t bytes) -> void* {
        void* r = (void*)p;
        p += (bytes + 255) & ~(size_t)255;
        return r;
    };
    bf16* WhT   = (bf16*)alloc((size_t)Hn * Hn * 2);   // W_h^T  [H][H]
    bf16* WimT  = (bf16*)alloc((size_t)Hn * Dn * 2);   // W_im^T [H][D]
    bf16* WoutT = (bf16*)alloc((size_t)Vn * Hn * 2);   // W_out^T[V][H]
    bf16* imgB  = (bf16*)alloc((size_t)Bn * Dn * 2);
    bf16* h0B   = (bf16*)alloc((size_t)Bn * Hn * 2);
    bf16* hB    = (bf16*)alloc((size_t)NSEQ * Hn * 2);
    bf16* hNewB = (bf16*)alloc((size_t)NSEQ * Hn * 2);
    float* imgProj = (float*)alloc((size_t)Bn * Hn * 4);
    float* h0New   = (float*)alloc((size_t)Bn * Hn * 4);
    float* pre     = (float*)alloc((size_t)NSEQ * Hn * 4);
    float* hNew    = (float*)alloc((size_t)NSEQ * Hn * 4);
    float* logits  = (float*)alloc((size_t)NSEQ * Vn * 4);
    float* lpA     = (float*)alloc((size_t)NSEQ * 4);
    float* lpB     = (float*)alloc((size_t)NSEQ * 4);
    float* topLp   = (float*)alloc((size_t)Bn * Sn * Sn * BEAMW * KSEL * 4);
    int*   topCls  = (int*)alloc((size_t)Bn * Sn * Sn * BEAMW * KSEL * 4);
    int*   preds   = (int*)alloc((size_t)Tn * NSEQ * 4);
    int*   bps     = (int*)alloc((size_t)Tn * NSEQ * 4);

    auto cvtT = [&](const float* s, bf16* d2, int R, int C) {
        dim3 grid((C + 31) / 32, (R + 31) / 32);
        cvt_bf16_transpose_kernel<<<grid, 256, 0, stream>>>(s, d2, R, C);
    };
    auto gemm = [&](const bf16* A, const bf16* Bt, float* C, int M, int K, int N) {
        int waves = (M / 16) * (N / 32);
        int blocks = (waves + 7) / 8;  // 8 waves (256 threads) per block
        gemm_bf16_wmma_kernel<<<blocks, 256, 0, stream>>>(A, Bt, C, M, K, N);
    };

    // one-time bf16 conversions; weights transposed so GEMM B-fragments are K-contiguous.
    // W_out^T is 20.5 MB bf16 -> L2-resident across all 20 steps on 192 MB L2.
    cvtT(W_h,   WhT,   Hn, Hn);
    cvtT(W_im,  WimT,  Dn, Hn);
    cvtT(W_out, WoutT, Hn, Vn);
    {
        long n = (long)Bn * Dn;
        cvt_bf16_kernel<<<(int)((n + 255) / 256), 256, 0, stream>>>(image_feature, imgB, n);
    }

    // img_proj[B,H] = image_feature @ W_im  (32x2048x512 WMMA)
    gemm(imgB, WimT, imgProj, Bn, Dn, Hn);

    // step 0: h0 = tanh(emb[start] + img_proj); lp0 = log_softmax(h0 @ W_out)
    step0_hidden_kernel<<<(Bn * Hn + 255) / 256, 256, 0, stream>>>(
        emb, imgProj, start_predictions, h0New, h0B);
    gemm(h0B, WoutT, logits, Bn, Hn, Vn);                     // 32x512x20000 WMMA
    log_softmax_kernel<<<Bn, 256, 0, stream>>>(logits, Vn);
    init_top5_kernel<<<Bn * Sn, 256, 0, stream>>>(logits, state_tr, lpA, preds);
    broadcast_h_kernel<<<(int)(((long)NSEQ * Hn + 255) / 256), 256, 0, stream>>>(h0New, hB);

    float* lpPrev = lpA;
    float* lpNext = lpB;
    for (int t = 1; t < Tn; ++t) {
        gemm(hB, WhT, pre, NSEQ, Hn, Hn);                     // 640x512x512 WMMA
        rnn_epilogue_kernel<<<(int)(((long)NSEQ * Hn + 255) / 256), 256, 0, stream>>>(
            pre, emb, imgProj, preds + (t - 1) * NSEQ, hNew, hNewB);
        gemm(hNewB, WoutT, logits, NSEQ, Hn, Vn);             // 640x512x20000 WMMA (dominant)
        log_softmax_kernel<<<NSEQ, 256, 0, stream>>>(logits, Vn);
        masked_top5_kernel<<<Bn * Sn * Sn * BEAMW, 256, 0, stream>>>(
            logits, state_tr, preds + (t - 1) * NSEQ, topLp, topCls);
        combine_kernel<<<Bn * Sn, 128, 0, stream>>>(
            topLp, topCls, lpPrev, lpNext, preds + t * NSEQ, bps + t * NSEQ);
        gather_h_kernel<<<NSEQ, 256, 0, stream>>>(hNew, bps + t * NSEQ, hB);
        float* tmp = lpPrev; lpPrev = lpNext; lpNext = tmp;
    }

    backtrack_kernel<<<(NSEQ + 255) / 256, 256, 0, stream>>>(preds, bps, lpPrev, out);
}
